// PaCNet_72550587564349
// MI455X (gfx1250) — matrix operand order repaired and checked by reading, not compiled, not run on hardware
//
#include <hip/hip_runtime.h>
#include <hip/hip_bf16.h>

typedef __attribute__((ext_vector_type(16))) __bf16 v16bf;
typedef __attribute__((ext_vector_type(8)))  float  v8f;

#define P 9216       // 96*96
#define VH 96
#define PT 576       // P/16 pixel tiles
#define MT 10        // ceil(150/16)
#define KT 5         // ceil(150/32)

__device__ __forceinline__ int refl(int q) {
    q = (q < 0) ? -q : q;
    return (q > 95) ? (190 - q) : q;
}

// Grouped 7x7 reflect conv (3in/3out per group); epilogue packs bf16 into WMMA
// B-tile layout: tile (g,pt,kt) is 512 contiguous bf16; within a tile
// lane = 16*(k_in_tile/16) + (p%16), elem j = k_in_tile%16.
__global__ __launch_bounds__(256) void conv7x7_pack(
    const float* __restrict__ in, const float* __restrict__ wvh,
    __bf16* __restrict__ bpack) {
    int ch = blockIdx.y;
    __shared__ float w[147];
    for (int i = threadIdx.x; i < 147; i += 256) w[i] = wvh[ch * 147 + i];
    __syncthreads();
    int p  = blockIdx.x * 256 + threadIdx.x;   // grid.x = 36 -> exact
    int px = p % VH, py = p / VH;
    int cibase = (ch / 3) * 3;
    float acc = 0.f;
    #pragma unroll
    for (int cl = 0; cl < 3; ++cl) {
        const float* inc = in + (size_t)(cibase + cl) * P;
        #pragma unroll
        for (int dy = 0; dy < 7; ++dy) {
            const float* row = inc + refl(py - 3 + dy) * VH;
            #pragma unroll
            for (int dx = 0; dx < 7; ++dx)
                acc += w[cl * 49 + dy * 7 + dx] * row[refl(px - 3 + dx)];
        }
    }
    int g = ch / 150, k = ch % 150;
    int kt = k >> 5, kk = k & 31, half = kk >> 4, j = kk & 15;
    int pt = p >> 4, ncol = p & 15, lane = half * 16 + ncol;
    bpack[((size_t)((g * PT + pt) * KT + kt)) * 512 + lane * 16 + j] = (__bf16)acc;
}

// Same conv, plain f32 output (block4).
__global__ __launch_bounds__(256) void conv7x7_f32(
    const float* __restrict__ in, const float* __restrict__ wvh,
    float* __restrict__ out) {
    int ch = blockIdx.y;
    __shared__ float w[147];
    for (int i = threadIdx.x; i < 147; i += 256) w[i] = wvh[ch * 147 + i];
    __syncthreads();
    int p  = blockIdx.x * 256 + threadIdx.x;
    int px = p % VH, py = p / VH;
    int cibase = (ch / 3) * 3;
    float acc = 0.f;
    #pragma unroll
    for (int cl = 0; cl < 3; ++cl) {
        const float* inc = in + (size_t)(cibase + cl) * P;
        #pragma unroll
        for (int dy = 0; dy < 7; ++dy) {
            const float* row = inc + refl(py - 3 + dy) * VH;
            #pragma unroll
            for (int dx = 0; dx < 7; ++dx)
                acc += w[cl * 49 + dy * 7 + dx] * row[refl(px - 3 + dx)];
        }
    }
    out[(size_t)ch * P + p] = acc;
}

// Pack per-neighbor 150x150 f32 weights into 16-bit A(16x32) WMMA layout,
// zero-padded to 160x160. Lanes 0-15 hold K {0..15}, lanes 16-31
// hold K {16..31} (2 packed per VGPR), matching the B pack.
__global__ __launch_bounds__(256) void apack(
    const float* __restrict__ wf, __bf16* __restrict__ ap, int total) {
    int idx = blockIdx.x * 256 + threadIdx.x;
    if (idx >= total) return;
    int j = idx & 15;
    int t1 = idx >> 4;
    int lane = t1 & 31;
    int t2 = t1 >> 5;
    int kt = t2 % KT;
    int t3 = t2 / KT;
    int mt = t3 % MT;
    int g  = t3 / MT;
    int m = mt * 16 + (lane & 15);
    int half = lane >> 4, pp = j >> 1, c = j & 1;
    int kin = half * 16 + 2 * pp + c;
    int k = kt * 32 + kin;
    float v = (m < 150 && k < 150) ? wf[(size_t)(g * 150 + m) * 150 + k] : 0.f;
    ap[idx] = (__bf16)v;
}

// One wave per 16x16 output tile; K=160 in 5 steps of v_wmma_f32_16x16x32_bf16.
__global__ __launch_bounds__(256) void gemm_wmma(
    const __bf16* __restrict__ Apk, const __bf16* __restrict__ Bpk,
    float* __restrict__ T2, int total_tiles) {
    int wave = threadIdx.x >> 5, lane = threadIdx.x & 31;
    int t = blockIdx.x * 8 + wave;
    if (t >= total_tiles) return;              // wave-uniform: EXEC stays all-1
    int pt = t % PT;
    int t2 = t / PT;
    int mt = t2 % MT;
    int g  = t2 / MT;
    const __bf16* Ab = Apk + ((size_t)((g * MT + mt) * KT)) * 512 + lane * 16;
    const __bf16* Bb = Bpk + ((size_t)((g * PT + pt) * KT)) * 512 + lane * 16;
    v8f acc = {};
    #pragma unroll
    for (int kt = 0; kt < KT; ++kt) {
        v16bf a = *(const v16bf*)(Ab + (size_t)kt * 512);
        v16bf b = *(const v16bf*)(Bb + (size_t)kt * 512);
        acc = __builtin_amdgcn_wmma_f32_16x16x32_bf16(
            false, a, false, b, (short)0, acc, false, false);
    }
    int half = lane >> 4, ncol = lane & 15;
    int p = pt * 16 + ncol;
    float* outp = T2 + (size_t)g * 150 * P + p;
    #pragma unroll
    for (int r = 0; r < 8; ++r) {
        int m = mt * 16 + half * 8 + r;
        if (m < 150) outp[(size_t)m * P] = acc[r];
    }
}

// neighbor mix + bias + relu (block0)
__global__ __launch_bounds__(256) void nmix_bias_relu(
    const float* __restrict__ T2, const float* __restrict__ wn,
    const float* __restrict__ bias, float* __restrict__ out,
    int n_in, int n_out) {
    int idx = blockIdx.x * 256 + threadIdx.x;
    int p = idx % P;
    int t = idx / P;
    int f  = t % 150;
    int no = t / 150;
    float acc = 0.f;
    for (int ni = 0; ni < n_in; ++ni)
        acc += wn[(size_t)(f * n_out + no) * n_in + ni] *
               T2[(size_t)(ni * 150 + f) * P + p];
    acc += bias[no * 150 + f];
    out[idx] = fmaxf(acc, 0.f);
}

// neighbor mix + eval BN + relu (blocks 1..3)
__global__ __launch_bounds__(256) void nmix_bn_relu(
    const float* __restrict__ T2, const float* __restrict__ wn,
    const float* __restrict__ gg, const float* __restrict__ be,
    const float* __restrict__ mm, const float* __restrict__ vv,
    float* __restrict__ out, int n_in, int n_out) {
    int idx = blockIdx.x * 256 + threadIdx.x;
    int p = idx % P;
    int t = idx / P;
    int f  = t % 150;
    int no = t / 150;
    float acc = 0.f;
    for (int ni = 0; ni < n_in; ++ni)
        acc += wn[(size_t)(f * n_out + no) * n_in + ni] *
               T2[(size_t)(ni * 150 + f) * P + p];
    int c = no * 150 + f;
    float inv = gg[c] * rsqrtf(vv[c] + 1e-5f);
    acc = acc * inv + (be[c] - mm[c] * inv);
    out[idx] = fmaxf(acc, 0.f);
}

// dense 1x1 150->3 + bias, then x_valid - z
__global__ __launch_bounds__(256) void final_out(
    const float* __restrict__ tbuf, const float* __restrict__ wf4,
    const float* __restrict__ b4, const float* __restrict__ xv,
    float* __restrict__ out) {
    int idx = blockIdx.x * 256 + threadIdx.x;
    if (idx >= 3 * P) return;
    int c = idx / P, p = idx % P;
    float z = b4[c];
    for (int k = 0; k < 150; ++k)
        z += wf4[c * 150 + k] * tbuf[(size_t)k * P + p];
    out[idx] = xv[idx] - z;
}

extern "C" void kernel_launch(void* const* d_in, const int* in_sizes, int n_in,
                              void* d_out, int out_size, void* d_ws, size_t ws_size,
                              hipStream_t stream) {
    const float* x_f     = (const float*)d_in[0];
    const float* x_valid = (const float*)d_in[2];
    const float* w_vh[5] = {(const float*)d_in[3], (const float*)d_in[7],
                            (const float*)d_in[14], (const float*)d_in[21],
                            (const float*)d_in[28]};
    const float* w_f[5]  = {(const float*)d_in[4], (const float*)d_in[8],
                            (const float*)d_in[15], (const float*)d_in[22],
                            (const float*)d_in[29]};
    const float* w_n[4]  = {(const float*)d_in[5], (const float*)d_in[9],
                            (const float*)d_in[16], (const float*)d_in[23]};
    const float* b0 = (const float*)d_in[6];
    const float* bn_g[4]  = {0, (const float*)d_in[10], (const float*)d_in[17], (const float*)d_in[24]};
    const float* bn_be[4] = {0, (const float*)d_in[11], (const float*)d_in[18], (const float*)d_in[25]};
    const float* bn_m[4]  = {0, (const float*)d_in[12], (const float*)d_in[19], (const float*)d_in[26]};
    const float* bn_v[4]  = {0, (const float*)d_in[13], (const float*)d_in[20], (const float*)d_in[27]};
    const float* b4 = (const float*)d_in[30];

    char* ws = (char*)d_ws;
    // workspace layout (bytes, 256-aligned)
    size_t off_A  = 0;                       // Apack  : <= 768 KB
    size_t off_B  = off_A + (1u << 20);      // Bpack  : <= 44,236,800
    size_t off_T  = off_B + 44236800u;       // T2     : <= 82,944,000
    size_t off_a1 = off_T + 82944000u;       // act1   : <= 44,236,800
    size_t off_a2 = off_a1 + 44236800u;      // act2   : <= 22,118,400
    __bf16* Apk = (__bf16*)(ws + off_A);
    __bf16* Bpk = (__bf16*)(ws + off_B);
    float*  T2  = (float*)(ws + off_T);
    float*  act1 = (float*)(ws + off_a1);
    float*  act2 = (float*)(ws + off_a2);

    const int n_ins[4]  = {15, 8, 4, 2};
    const int n_outs[4] = {8, 4, 2, 1};
    const float* cur = x_f;
    float* outs[4] = {act1, act2, act1, act2};

    for (int blk = 0; blk < 4; ++blk) {
        int G = n_ins[blk], NO = n_outs[blk];
        int C = G * 150;
        conv7x7_pack<<<dim3(36, C), 256, 0, stream>>>(cur, w_vh[blk], Bpk);
        int totA = G * MT * KT * 512;
        apack<<<(totA + 255) / 256, 256, 0, stream>>>(w_f[blk], Apk, totA);
        int tiles = G * MT * PT;
        gemm_wmma<<<(tiles + 7) / 8, 256, 0, stream>>>(Apk, Bpk, T2, tiles);
        int totO = NO * 150 * P;
        if (blk == 0)
            nmix_bias_relu<<<(totO + 255) / 256, 256, 0, stream>>>(
                T2, w_n[0], b0, outs[0], G, NO);
        else
            nmix_bn_relu<<<(totO + 255) / 256, 256, 0, stream>>>(
                T2, w_n[blk], bn_g[blk], bn_be[blk], bn_m[blk], bn_v[blk],
                outs[blk], G, NO);
        cur = outs[blk];
    }

    // block4: grouped 7x7 on (150,96,96) -> T2 region, then dense 1x1 + residual
    conv7x7_f32<<<dim3(36, 150), 256, 0, stream>>>(cur, w_vh[4], T2);
    final_out<<<(3 * P + 255) / 256, 256, 0, stream>>>(
        T2, w_f[4], b4, x_valid, (float*)d_out);
}
